// TransformerPPAttention_41678362640852
// MI455X (gfx1250) — matrix operand order repaired and checked by reading, not compiled, not run on hardware
//
#include <hip/hip_runtime.h>

// =====================================================================
// Transformer attention block (qkv proj + RoPE + causal flash attention
// + out proj) for MI455X / gfx1250.  All matmuls use
// v_wmma_f32_16x16x32_bf16 (f32 accumulate); bf16 staging via LDS;
// pure-copy stages use GLOBAL_LOAD_ASYNC_TO_LDS_B128 (ASYNCcnt).
// =====================================================================

typedef __attribute__((ext_vector_type(16))) __bf16 v16bf;
typedef __attribute__((ext_vector_type(2)))  __bf16 v2bf;
typedef __attribute__((ext_vector_type(8)))  float  v8f;

union Frag16 {
  v16bf v;
  uint4 q2[2];
};

__device__ __forceinline__ unsigned short f2bf(float f) {
  __bf16 h = (__bf16)f;
  return __builtin_bit_cast(unsigned short, h);
}

#if __has_builtin(__builtin_amdgcn_cvt_pk_bf16_f32)
__device__ __forceinline__ unsigned pack2(float a, float b) {
  v2bf t = __builtin_amdgcn_cvt_pk_bf16_f32(a, b);
  return __builtin_bit_cast(unsigned, t);
}
#else
__device__ __forceinline__ unsigned pack2(float a, float b) {
  v2bf t; t[0] = (__bf16)a; t[1] = (__bf16)b;
  return __builtin_bit_cast(unsigned, t);
}
#endif

// Async global->LDS copy of 16 bytes per lane (CDNA5 ASYNCcnt path).
__device__ __forceinline__ void async_copy16(unsigned ldsOff, const void* gptr) {
  asm volatile("global_load_async_to_lds_b128 %0, %1, off"
               :: "v"(ldsOff), "v"(gptr) : "memory");
}
__device__ __forceinline__ void wait_async() {
#if __has_builtin(__builtin_amdgcn_s_wait_asynccnt)
  __builtin_amdgcn_s_wait_asynccnt(0);
#else
  asm volatile("s_wait_asynccnt 0" ::: "memory");
#endif
}

#define LDT 40  // padded LDS row stride (bf16 units) for GEMM tiles

// ---------------------------------------------------------------------
// Kernel 1: qkv = x @ qkv_w.T + b, fused RoPE on q/k, scatter to
// [B][nh][S][hd] bf16 workspaces.  M=4096, N=6144, K=2048.
// Block tile 128x128, 256 threads (8 waves as 4Mx2N), wave tile 32x64.
// ---------------------------------------------------------------------
__global__ __launch_bounds__(256) void qkv_rope_kernel(
    const float* __restrict__ x, const float* __restrict__ w,
    const float* __restrict__ bias,
    unsigned short* __restrict__ q_ws, unsigned short* __restrict__ k_ws,
    unsigned short* __restrict__ v_ws)
{
  __shared__ __align__(16) unsigned short Als[128 * LDT];
  __shared__ __align__(16) unsigned short Bls[128 * LDT];

  const int tM = blockIdx.y;   // 0..31
  const int tN = blockIdx.x;   // 0..47
  const int tid  = threadIdx.x;
  const int wave = tid >> 5;
  const int lane = tid & 31;
  const int wm = wave >> 1;    // 0..3
  const int wn = wave & 1;     // 0..1
  const int r  = lane & 15;
  const int hh = lane >> 4;

  const int srow = tid >> 1;           // staging row 0..127
  const int scol = (tid & 1) * 16;     // staging col base

  const float* xrow = x + (size_t)(tM * 128 + srow) * 2048 + scol;
  const float* wrow = w + (size_t)(tN * 128 + srow) * 2048 + scol;
  unsigned short* aSt = &Als[srow * LDT + scol];
  unsigned short* bSt = &Bls[srow * LDT + scol];

  v8f acc[2][4] = {};

  for (int kk = 0; kk < 2048; kk += 32) {
    __syncthreads();
#pragma unroll
    for (int i = 0; i < 4; ++i) {
      float4 av = *(const float4*)(xrow + kk + i * 4);
      float4 bv = *(const float4*)(wrow + kk + i * 4);
      uint2 ap; ap.x = pack2(av.x, av.y); ap.y = pack2(av.z, av.w);
      uint2 bp; bp.x = pack2(bv.x, bv.y); bp.y = pack2(bv.z, bv.w);
      *(uint2*)(aSt + i * 4) = ap;
      *(uint2*)(bSt + i * 4) = bp;
    }
    if (kk + 32 < 2048) {
      __builtin_prefetch(xrow + kk + 32, 0, 0);
      __builtin_prefetch(wrow + kk + 32, 0, 0);
    }
    __syncthreads();

    Frag16 fa[2], fb[4];
#pragma unroll
    for (int mi = 0; mi < 2; ++mi) {
      const unsigned short* p = &Als[(wm * 32 + mi * 16 + r) * LDT + hh * 8];
      fa[mi].q2[0] = *(const uint4*)p;
      fa[mi].q2[1] = *(const uint4*)(p + 16);
    }
#pragma unroll
    for (int ni = 0; ni < 4; ++ni) {
      const unsigned short* p = &Bls[(wn * 64 + ni * 16 + r) * LDT + hh * 8];
      fb[ni].q2[0] = *(const uint4*)p;
      fb[ni].q2[1] = *(const uint4*)(p + 16);
    }
#pragma unroll
    for (int mi = 0; mi < 2; ++mi)
#pragma unroll
      for (int ni = 0; ni < 4; ++ni)
        acc[mi][ni] = __builtin_amdgcn_wmma_f32_16x16x32_bf16(
            false, fa[mi].v, false, fb[ni].v, (short)0, acc[mi][ni], false, false);
  }

  // Epilogue: bias + RoPE (interleaved rotate-half, concat freq tables)
#pragma unroll
  for (int mi = 0; mi < 2; ++mi)
#pragma unroll
    for (int ni = 0; ni < 4; ++ni) {
      const int gn = tN * 128 + wn * 64 + ni * 16 + r;
      const float bv = bias[gn];
      const int which = gn >> 11;       // gn / 2048: 0=q 1=k 2=v
      const int hn = gn & 2047;
      const int head = hn >> 7;
      const int d = hn & 127;
#pragma unroll
      for (int j = 0; j < 8; ++j) {
        const int gm = tM * 128 + wm * 32 + mi * 16 + j + 8 * hh;
        const int b = gm >> 11;
        const int s = gm & 2047;
        float val = acc[mi][ni][j] + bv;
        const size_t dstIdx = (((size_t)(b * 16 + head)) * 2048 + s) * 128 + d;
        if (which == 2) {
          v_ws[dstIdx] = f2bf(val);
        } else {
          // partner lane holds the paired element d^1 of the same row
          float p = __shfl_xor(val, 1, 32);
          const int fi = d & 63;
          const float ang = (float)s * __powf(10000.0f, -(float)fi * (1.0f / 64.0f));
          float sn, cs;
          __sincosf(ang, &sn, &cs);
          const float rot = (d & 1) ? p : -p;
          const float o = val * cs + rot * sn;
          unsigned short* dst = (which == 0) ? q_ws : k_ws;
          dst[dstIdx] = f2bf(o);
        }
      }
    }
}

// ---------------------------------------------------------------------
// Kernel 2: causal flash attention per (b, head, 64-query tile).
// 128 threads = 4 waves; each wave owns 16 query rows.
// K-tile staged via async global->LDS (ASYNCcnt); V staged transposed
// so the P@V B-operand loads with the canonical row-major [N][K] pattern.
// ---------------------------------------------------------------------
__global__ __launch_bounds__(128) void flash_attn_kernel(
    const unsigned short* __restrict__ q_ws,
    const unsigned short* __restrict__ k_ws,
    const unsigned short* __restrict__ v_ws,
    const int* __restrict__ amask,
    unsigned short* __restrict__ attn_ws)
{
  __shared__ __align__(16) unsigned short Kls[64 * 136];   // [key][hd], pad 8
  __shared__ __align__(16) unsigned short Vt[128 * 72];    // [hd][key], pad 8
  __shared__ __align__(16) unsigned short Pls[4 * 16 * 72];// per-wave P tiles

  const int qt   = blockIdx.x;   // 0..31
  const int head = blockIdx.y;   // 0..15
  const int b    = blockIdx.z;   // 0..1
  const int tid  = threadIdx.x;
  const int wave = tid >> 5;
  const int lane = tid & 31;
  const int r  = lane & 15;
  const int hh = lane >> 4;
  const int q0 = qt * 64;

  const size_t baseBH = ((size_t)(b * 16 + head)) * 2048 * 128;

  // Resident Q fragments (16 rows x 128 hd): 4 A-frags
  Frag16 fq[4];
  {
    const unsigned short* qp = q_ws + baseBH + (size_t)(q0 + wave * 16 + r) * 128;
#pragma unroll
    for (int kb = 0; kb < 4; ++kb) {
      fq[kb].q2[0] = *(const uint4*)(qp + kb * 32 + hh * 8);
      fq[kb].q2[1] = *(const uint4*)(qp + kb * 32 + 16 + hh * 8);
    }
  }

  v8f o[8] = {};
  float m_r[8], l_r[8];
#pragma unroll
  for (int j = 0; j < 8; ++j) { m_r[j] = -__builtin_inff(); l_r[j] = 0.0f; }

  const float scale = 0.08838834764831845f;  // 128^-0.5
  const int myrow0 = q0 + wave * 16 + 8 * hh;

  for (int kt = 0; kt <= qt; ++kt) {
    const int k0 = kt * 64;
    __syncthreads();
    {
      // Stage K rows via async global->LDS; V transposed via VGPR path.
      const int row = tid >> 1;
      const int ch = (tid & 1) * 64;
      const unsigned short* ksrc = k_ws + baseBH + (size_t)(k0 + row) * 128 + ch;
      unsigned short* kdst = &Kls[row * 136 + ch];
#pragma unroll
      for (int i = 0; i < 8; ++i)
        async_copy16((unsigned)(uintptr_t)(kdst + i * 8), ksrc + i * 8);
      const unsigned short* vsrc = v_ws + baseBH + (size_t)(k0 + row) * 128 + ch;
#pragma unroll
      for (int i = 0; i < 64; ++i) Vt[(ch + i) * 72 + row] = vsrc[i];
    }
    wait_async();
    __syncthreads();

    // scores: 16x64 per wave
    v8f sc[4] = {};
#pragma unroll
    for (int n = 0; n < 4; ++n)
#pragma unroll
      for (int kb = 0; kb < 4; ++kb) {
        Frag16 fk;
        const unsigned short* kp = &Kls[(n * 16 + r) * 136 + kb * 32 + hh * 8];
        fk.q2[0] = *(const uint4*)kp;
        fk.q2[1] = *(const uint4*)(kp + 16);
        sc[n] = __builtin_amdgcn_wmma_f32_16x16x32_bf16(
            false, fq[kb].v, false, fk.v, (short)0, sc[n], false, false);
      }

    // scale + causal + key mask
#pragma unroll
    for (int n = 0; n < 4; ++n) {
      const int key = k0 + n * 16 + r;
      const bool kvalid = (amask[b * 2048 + key] != 0);
#pragma unroll
      for (int j = 0; j < 8; ++j) {
        float v = sc[n][j] * scale;
        if (!kvalid || key > (myrow0 + j)) v = -__builtin_inff();
        sc[n][j] = v;
      }
    }

    // online softmax per row (row lives in 16 lanes of one half)
#pragma unroll
    for (int j = 0; j < 8; ++j) {
      float mx = fmaxf(fmaxf(sc[0][j], sc[1][j]), fmaxf(sc[2][j], sc[3][j]));
#pragma unroll
      for (int off = 1; off < 16; off <<= 1) mx = fmaxf(mx, __shfl_xor(mx, off, 32));
      const float mnew = fmaxf(m_r[j], mx);
      const float alpha = __expf(m_r[j] - mnew);
      float rs = 0.0f;
#pragma unroll
      for (int n = 0; n < 4; ++n) {
        float p = __expf(sc[n][j] - mnew);
        sc[n][j] = p;
        rs += p;
      }
#pragma unroll
      for (int off = 1; off < 16; off <<= 1) rs += __shfl_xor(rs, off, 32);
      l_r[j] = l_r[j] * alpha + rs;
      m_r[j] = mnew;
#pragma unroll
      for (int no = 0; no < 8; ++no) o[no][j] = o[no][j] * alpha;
    }

    // C-layout P -> LDS -> A-layout (wave-private region, DS in-order)
    unsigned short* pbase = &Pls[wave * 16 * 72];
#pragma unroll
    for (int n = 0; n < 4; ++n)
#pragma unroll
      for (int j = 0; j < 8; ++j)
        pbase[(j + 8 * hh) * 72 + n * 16 + r] = f2bf(sc[n][j]);

#pragma unroll
    for (int kb2 = 0; kb2 < 2; ++kb2) {
      Frag16 fp;
      const unsigned short* pp = &pbase[r * 72 + kb2 * 32 + hh * 8];
      fp.q2[0] = *(const uint4*)pp;
      fp.q2[1] = *(const uint4*)(pp + 16);
#pragma unroll
      for (int no = 0; no < 8; ++no) {
        Frag16 fv;
        const unsigned short* vp = &Vt[(no * 16 + r) * 72 + kb2 * 32 + hh * 8];
        fv.q2[0] = *(const uint4*)vp;
        fv.q2[1] = *(const uint4*)(vp + 16);
        o[no] = __builtin_amdgcn_wmma_f32_16x16x32_bf16(
            false, fp.v, false, fv.v, (short)0, o[no], false, false);
      }
    }
  }

  // epilogue: normalize and scatter to [B][S][H] bf16
  const size_t outBase = ((size_t)b * 2048) * 2048 + head * 128;
#pragma unroll
  for (int no = 0; no < 8; ++no)
#pragma unroll
    for (int j = 0; j < 8; ++j) {
      const int row = q0 + wave * 16 + j + 8 * hh;
      const float val = o[no][j] / l_r[j];
      attn_ws[outBase + (size_t)row * 2048 + no * 16 + r] = f2bf(val);
    }
}

// ---------------------------------------------------------------------
// Kernel 3: out = attn @ out_w.T + out_b.  M=4096, N=2048, K=2048.
// A tile staged via async global->LDS; B converted f32->bf16 in flight.
// ---------------------------------------------------------------------
__global__ __launch_bounds__(256) void out_proj_kernel(
    const unsigned short* __restrict__ attn, const float* __restrict__ w,
    const float* __restrict__ bias, float* __restrict__ out)
{
  __shared__ __align__(16) unsigned short Als[128 * LDT];
  __shared__ __align__(16) unsigned short Bls[128 * LDT];

  const int tM = blockIdx.y;   // 0..31
  const int tN = blockIdx.x;   // 0..15
  const int tid  = threadIdx.x;
  const int wave = tid >> 5;
  const int lane = tid & 31;
  const int wm = wave >> 1;
  const int wn = wave & 1;
  const int r  = lane & 15;
  const int hh = lane >> 4;

  const int srow = tid >> 1;
  const int scol = (tid & 1) * 16;

  const unsigned short* arow = attn + (size_t)(tM * 128 + srow) * 2048 + scol;
  const float* wrow = w + (size_t)(tN * 128 + srow) * 2048 + scol;
  unsigned short* aSt = &Als[srow * LDT + scol];
  unsigned short* bSt = &Bls[srow * LDT + scol];

  v8f acc[2][4] = {};

  for (int kk = 0; kk < 2048; kk += 32) {
    __syncthreads();
    async_copy16((unsigned)(uintptr_t)aSt, arow + kk);
    async_copy16((unsigned)(uintptr_t)(aSt + 8), arow + kk + 8);
#pragma unroll
    for (int i = 0; i < 4; ++i) {
      float4 bv = *(const float4*)(wrow + kk + i * 4);
      uint2 bp; bp.x = pack2(bv.x, bv.y); bp.y = pack2(bv.z, bv.w);
      *(uint2*)(bSt + i * 4) = bp;
    }
    wait_async();
    __syncthreads();

    Frag16 fa[2], fb[4];
#pragma unroll
    for (int mi = 0; mi < 2; ++mi) {
      const unsigned short* p = &Als[(wm * 32 + mi * 16 + r) * LDT + hh * 8];
      fa[mi].q2[0] = *(const uint4*)p;
      fa[mi].q2[1] = *(const uint4*)(p + 16);
    }
#pragma unroll
    for (int ni = 0; ni < 4; ++ni) {
      const unsigned short* p = &Bls[(wn * 64 + ni * 16 + r) * LDT + hh * 8];
      fb[ni].q2[0] = *(const uint4*)p;
      fb[ni].q2[1] = *(const uint4*)(p + 16);
    }
#pragma unroll
    for (int mi = 0; mi < 2; ++mi)
#pragma unroll
      for (int ni = 0; ni < 4; ++ni)
        acc[mi][ni] = __builtin_amdgcn_wmma_f32_16x16x32_bf16(
            false, fa[mi].v, false, fb[ni].v, (short)0, acc[mi][ni], false, false);
  }

#pragma unroll
  for (int mi = 0; mi < 2; ++mi)
#pragma unroll
    for (int ni = 0; ni < 4; ++ni) {
      const int gn = tN * 128 + wn * 64 + ni * 16 + r;
      const float bv = bias[gn];
#pragma unroll
      for (int j = 0; j < 8; ++j) {
        const int gm = tM * 128 + wm * 32 + mi * 16 + j + 8 * hh;
        out[(size_t)gm * 2048 + gn] = acc[mi][ni][j] + bv;
      }
    }
}

// ---------------------------------------------------------------------
extern "C" void kernel_launch(void* const* d_in, const int* in_sizes, int n_in,
                              void* d_out, int out_size, void* d_ws, size_t ws_size,
                              hipStream_t stream) {
  (void)in_sizes; (void)n_in; (void)out_size; (void)ws_size;

  const float* x      = (const float*)d_in[0];
  const int*   amask  = (const int*)d_in[1];
  const float* qkv_w  = (const float*)d_in[2];
  const float* qkv_b  = (const float*)d_in[3];
  const float* out_w  = (const float*)d_in[4];
  const float* out_b  = (const float*)d_in[5];
  float* out = (float*)d_out;

  const size_t QKV_ELEMS = (size_t)2 * 16 * 2048 * 128;  // 8.4M bf16 each
  unsigned short* q_ws    = (unsigned short*)d_ws;
  unsigned short* k_ws    = q_ws + QKV_ELEMS;
  unsigned short* v_ws    = k_ws + QKV_ELEMS;
  unsigned short* attn_ws = v_ws + QKV_ELEMS;            // 64 MB total

  qkv_rope_kernel<<<dim3(48, 32), 256, 0, stream>>>(x, qkv_w, qkv_b, q_ws, k_ws, v_ws);
  flash_attn_kernel<<<dim3(32, 16, 2), 128, 0, stream>>>(q_ws, k_ws, v_ws, amask, attn_ws);
  out_proj_kernel<<<dim3(16, 32), 256, 0, stream>>>(attn_ws, out_w, out_b, out);
}
